// Attention_Sim_38062000177267
// MI455X (gfx1250) — matrix operand order repaired and checked by reading, not compiled, not run on hardware
//
#include <hip/hip_runtime.h>
#include <hip/hip_bf16.h>

typedef __attribute__((ext_vector_type(16))) _Float16 v16h;
typedef __attribute__((ext_vector_type(8)))  _Float16 v8h;
typedef __attribute__((ext_vector_type(8)))  float    v8f;

union ABFrag { v16h v; v8h h[2]; };

__device__ __forceinline__ v8f wmma16(const ABFrag& a, const ABFrag& b, v8f c) {
    // D = A(16x32 f16) * B(32x16 f16) + C(16x16 f32)
    return __builtin_amdgcn_wmma_f32_16x16x32_f16(
        /*neg_a=*/false, a.v, /*neg_b=*/false, b.v,
        /*c_mod=*/(short)0, c, /*reuse_a=*/false, /*reuse_b=*/false);
}

// A-fragment: lane = M row, halfs = K {off..off+7, off+8..off+15}
__device__ __forceinline__ void load_afrag(ABFrag& f, const _Float16* row, int off) {
    f.h[0] = *(const v8h*)(row + off);
    f.h[1] = *(const v8h*)(row + off + 8);
}
// B-fragment from [N][K] storage: row ptr already includes n*stride + 16*hi
__device__ __forceinline__ void load_bfrag(ABFrag& f, const _Float16* row, int off) {
    f.h[0] = *(const v8h*)(row + off);
    f.h[1] = *(const v8h*)(row + off + 8);
}

// Problem constants
#define BB 4
#define SS 1024
#define DD 1024
#define HH 16
#define HD 64

static const size_t XH_N  = (size_t)BB * SS * DD;      // 4,194,304
static const size_t WAT_N = (size_t)DD * 3 * DD;       // 3,145,728  ([3D][D] transposed)
static const size_t WPT_N = (size_t)DD * DD;           // 1,048,576  ([D][D] transposed)
static const size_t QKV_N = (size_t)BB * HH * SS * HD; // 4,194,304 each

// ---------------------------------------------------------------------------
// Kernel 1: fp32 -> f16 conversion; weights transposed to [N][K] so WMMA
// B-fragments become contiguous 16B loads.
// ---------------------------------------------------------------------------
__global__ void prep_kernel(const float* __restrict__ x,
                            const float* __restrict__ w_attn,
                            const float* __restrict__ w_proj,
                            _Float16* __restrict__ xh,
                            _Float16* __restrict__ wat,
                            _Float16* __restrict__ wpt) {
    size_t idx = (size_t)blockIdx.x * blockDim.x + threadIdx.x;
    if (idx < XH_N) {
        xh[idx] = (_Float16)x[idx];
    } else if (idx < XH_N + WAT_N) {
        size_t i = idx - XH_N;
        int n = (int)(i >> 10);
        int k = (int)(i & 1023);
        wat[i] = (_Float16)w_attn[(size_t)k * (3 * DD) + n];
    } else if (idx < XH_N + WAT_N + WPT_N) {
        size_t i = idx - XH_N - WAT_N;
        int n = (int)(i >> 10);
        int k = (int)(i & 1023);
        wpt[i] = (_Float16)w_proj[(size_t)k * DD + n];
    }
}

// ---------------------------------------------------------------------------
// Kernel 2: QKV GEMM  [4096,1024] x [1024,3072] + bias, WMMA f16->f32.
// Each wave: one 32x64 output tile (2 m-subtiles x 4 n-subtiles = 8 accums).
// Per k-step: load all fragments into fresh locals (clause-batched b128s),
// then 8 WMMAs; unroll-by-2 lets the scheduler overlap iterations without
// forced copies (no spills, no v_mov/v_nop chains).
// Scatter: Q,K -> [B,H,S,64] f16 ; V -> [B,H,64,S] f16 (transposed).
// ---------------------------------------------------------------------------
__global__ void qkv_gemm_kernel(const _Float16* __restrict__ xh,
                                const _Float16* __restrict__ wat,   // [3072][1024]
                                const float* __restrict__ b_attn,
                                _Float16* __restrict__ Qh,
                                _Float16* __restrict__ Kh,
                                _Float16* __restrict__ Vt) {
    const int NT = (3 * DD) / 64;                         // 48 n-tiles
    int gwave = (blockIdx.x * blockDim.x + threadIdx.x) >> 5;
    int lane  = threadIdx.x & 31;
    int ntile = gwave % NT;
    int mtile = gwave / NT;
    int m0 = mtile * 32;
    int n0 = ntile * 64;
    int hi  = lane >> 4;
    int l15 = lane & 15;
    int akb = hi ? 16 : 0;

    v8f acc[2][4] = {};
    const _Float16* arow0 = xh + (size_t)(m0 + l15) * DD + akb;
    const _Float16* arow1 = arow0 + (size_t)16 * DD;
    const _Float16* brow[4];
#pragma unroll
    for (int t = 0; t < 4; ++t)
        brow[t] = wat + (size_t)(n0 + t * 16 + l15) * DD + 16 * hi;

#pragma unroll 2
    for (int ks = 0; ks < DD; ks += 32) {
        ABFrag a0, a1, b[4];
        load_afrag(a0, arow0, ks);
        load_afrag(a1, arow1, ks);
#pragma unroll
        for (int t = 0; t < 4; ++t) load_bfrag(b[t], brow[t], ks);
        __builtin_prefetch(arow0 + ks + 128, 0, 3);
#pragma unroll
        for (int t = 0; t < 4; ++t) {
            acc[0][t] = wmma16(a0, b[t], acc[0][t]);
            acc[1][t] = wmma16(a1, b[t], acc[1][t]);
        }
    }

    int region = n0 >> 10;          // 0=Q 1=K 2=V
    int nin    = n0 & 1023;
    int h      = nin >> 6;          // head index (64-wide tile == one head)

#pragma unroll
    for (int ms = 0; ms < 2; ++ms) {
#pragma unroll
        for (int t = 0; t < 4; ++t) {
#pragma unroll
            for (int r = 0; r < 8; ++r) {
                int m = m0 + ms * 16 + r + 8 * hi;       // token index
                int b2 = m >> 10, s = m & 1023;
                int d = t * 16 + l15;
                float val = acc[ms][t][r] + b_attn[n0 + t * 16 + l15];
                _Float16 hv = (_Float16)val;
                size_t bh = (size_t)b2 * HH + h;
                if (region == 0)       Qh[(bh * SS + s) * HD + d] = hv;
                else if (region == 1)  Kh[(bh * SS + s) * HD + d] = hv;
                else                   Vt[(bh * HD + d) * SS + s] = hv;
            }
        }
    }
}

// ---------------------------------------------------------------------------
// Kernel 3: causal flash attention. One wave per 16-row q tile.
// Key blocks of 32: 4 score WMMAs + online softmax + 4 P*V WMMAs.
// ---------------------------------------------------------------------------
__global__ void attn_kernel(const _Float16* __restrict__ Qh,
                            const _Float16* __restrict__ Kh,
                            const _Float16* __restrict__ Vt,
                            _Float16* __restrict__ ctxh) {
    __shared__ __align__(16) _Float16 plds[4][16 * 32];   // per-wave P scratch

    int wid  = threadIdx.x >> 5;
    int lane = threadIdx.x & 31;
    int gwave = blockIdx.x * 4 + wid;
    int bh = gwave >> 6;              // 64 q-tiles per (b,h)
    int qt = gwave & 63;
    int qbase = qt * 16;

    const _Float16* Qbh = Qh + (size_t)bh * SS * HD;
    const _Float16* Kbh = Kh + (size_t)bh * SS * HD;
    const _Float16* Vbh = Vt + (size_t)bh * HD * SS;

    int hi  = lane >> 4;
    int l15 = lane & 15;
    int kb  = hi ? 16 : 0;

    // Q A-fragments for K=0..31 and K=32..63
    ABFrag qa[2];
    const _Float16* qrow = Qbh + (size_t)(qbase + l15) * HD;
#pragma unroll
    for (int kk = 0; kk < 2; ++kk) load_afrag(qa[kk], qrow + kk * 32, kb);

    float mrow[8], lrow[8];
    v8f o[4] = {};
#pragma unroll
    for (int r = 0; r < 8; ++r) { mrow[r] = -1e30f; lrow[r] = 0.0f; }

    _Float16* pl = &plds[wid][0];
    int nblocks = (qbase + 16 + 31) >> 5;   // causal: keys <= qbase+15

    for (int j = 0; j < nblocks; ++j) {
        int jb = j * 32;
        if (j + 1 < nblocks) {
            // 32 lanes x 128B rows cover the whole next K block
            __builtin_prefetch(Kbh + (size_t)(jb + 32 + lane) * HD, 0, 3);
        }
        v8f sc[2] = {};
#pragma unroll
        for (int half = 0; half < 2; ++half) {
            const _Float16* krow = Kbh + (size_t)(jb + half * 16 + l15) * HD + 16 * hi;
            ABFrag b0, b1;
            load_bfrag(b0, krow, 0);
            load_bfrag(b1, krow, 32);
            sc[half] = wmma16(qa[0], b0, sc[half]);
            sc[half] = wmma16(qa[1], b1, sc[half]);
        }

        float p0s[8], p1s[8];
#pragma unroll
        for (int r = 0; r < 8; ++r) {
            int q = qbase + r + 8 * hi;
            float s0 = sc[0][r] * 0.125f;           // 1/sqrt(64)
            float s1 = sc[1][r] * 0.125f;
            if (jb + l15 > q)       s0 = -1e9f;
            if (jb + 16 + l15 > q)  s1 = -1e9f;
            float mx = fmaxf(s0, s1);
            mx = fmaxf(mx, __shfl_xor(mx, 1));
            mx = fmaxf(mx, __shfl_xor(mx, 2));
            mx = fmaxf(mx, __shfl_xor(mx, 4));
            mx = fmaxf(mx, __shfl_xor(mx, 8));
            float mnew = fmaxf(mrow[r], mx);
            float p0 = __expf(s0 - mnew);
            float p1 = __expf(s1 - mnew);
            float rs = p0 + p1;
            rs += __shfl_xor(rs, 1);
            rs += __shfl_xor(rs, 2);
            rs += __shfl_xor(rs, 4);
            rs += __shfl_xor(rs, 8);
            float corr = __expf(mrow[r] - mnew);
            lrow[r] = lrow[r] * corr + rs;
            mrow[r] = mnew;
            o[0][r] *= corr; o[1][r] *= corr; o[2][r] *= corr; o[3][r] *= corr;
            p0s[r] = p0; p1s[r] = p1;
        }

        // C-layout -> LDS [16][32] (row = r + 8*hi, col = half*16 + l15)
#pragma unroll
        for (int r = 0; r < 8; ++r) {
            pl[(r + 8 * hi) * 32 + l15]      = (_Float16)p0s[r];
            pl[(r + 8 * hi) * 32 + 16 + l15] = (_Float16)p1s[r];
        }
        // read back as A-fragment (wave-private; compiler inserts ds waits)
        ABFrag pa;
        const _Float16* prow = pl + l15 * 32;
        load_afrag(pa, prow, kb);

#pragma unroll
        for (int t = 0; t < 4; ++t) {
            const _Float16* vrow = Vbh + (size_t)(t * 16 + l15) * SS + jb + 16 * hi;
            ABFrag vb;
            load_bfrag(vb, vrow, 0);
            o[t] = wmma16(pa, vb, o[t]);
        }
    }

    // normalize and store merged-head context (f16)
    int b = bh >> 4, h = bh & 15;
#pragma unroll
    for (int t = 0; t < 4; ++t) {
#pragma unroll
        for (int r = 0; r < 8; ++r) {
            int s = qbase + r + 8 * hi;
            float val = o[t][r] / lrow[r];
            ctxh[((size_t)b * SS + s) * DD + h * HD + t * 16 + l15] = (_Float16)val;
        }
    }
}

// ---------------------------------------------------------------------------
// Kernel 4: output projection  [4096,1024] x [1024,1024] + bias -> fp32 out
// Same 32x64 wave tile as the QKV GEMM.
// ---------------------------------------------------------------------------
__global__ void proj_gemm_kernel(const _Float16* __restrict__ ctxh,
                                 const _Float16* __restrict__ wpt,  // [1024][1024]
                                 const float* __restrict__ b_proj,
                                 float* __restrict__ out) {
    const int NT = DD / 64;                               // 16 n-tiles
    int gwave = (blockIdx.x * blockDim.x + threadIdx.x) >> 5;
    int lane  = threadIdx.x & 31;
    int ntile = gwave % NT;
    int mtile = gwave / NT;
    int m0 = mtile * 32;
    int n0 = ntile * 64;
    int hi  = lane >> 4;
    int l15 = lane & 15;
    int akb = hi ? 16 : 0;

    v8f acc[2][4] = {};
    const _Float16* arow0 = ctxh + (size_t)(m0 + l15) * DD + akb;
    const _Float16* arow1 = arow0 + (size_t)16 * DD;
    const _Float16* brow[4];
#pragma unroll
    for (int t = 0; t < 4; ++t)
        brow[t] = wpt + (size_t)(n0 + t * 16 + l15) * DD + 16 * hi;

#pragma unroll 2
    for (int ks = 0; ks < DD; ks += 32) {
        ABFrag a0, a1, b[4];
        load_afrag(a0, arow0, ks);
        load_afrag(a1, arow1, ks);
#pragma unroll
        for (int t = 0; t < 4; ++t) load_bfrag(b[t], brow[t], ks);
        __builtin_prefetch(arow0 + ks + 128, 0, 3);
#pragma unroll
        for (int t = 0; t < 4; ++t) {
            acc[0][t] = wmma16(a0, b[t], acc[0][t]);
            acc[1][t] = wmma16(a1, b[t], acc[1][t]);
        }
    }

#pragma unroll
    for (int ms = 0; ms < 2; ++ms) {
#pragma unroll
        for (int t = 0; t < 4; ++t) {
#pragma unroll
            for (int r = 0; r < 8; ++r) {
                int m = m0 + ms * 16 + r + 8 * hi;
                int col = n0 + t * 16 + l15;
                out[(size_t)m * DD + col] = acc[ms][t][r] + b_proj[col];
            }
        }
    }
}

// ---------------------------------------------------------------------------
extern "C" void kernel_launch(void* const* d_in, const int* in_sizes, int n_in,
                              void* d_out, int out_size, void* d_ws, size_t ws_size,
                              hipStream_t stream) {
    const float* x      = (const float*)d_in[0];
    const float* w_attn = (const float*)d_in[1];
    const float* b_attn = (const float*)d_in[2];
    const float* w_proj = (const float*)d_in[3];
    const float* b_proj = (const float*)d_in[4];
    float* out = (float*)d_out;

    _Float16* xh   = (_Float16*)d_ws;
    _Float16* wat  = xh  + XH_N;
    _Float16* wpt  = wat + WAT_N;
    _Float16* Qh   = wpt + WPT_N;
    _Float16* Kh   = Qh  + QKV_N;
    _Float16* Vt   = Kh  + QKV_N;
    _Float16* ctxh = Vt  + QKV_N;
    (void)in_sizes; (void)n_in; (void)out_size; (void)ws_size;

    // 1) convert / transpose inputs to f16
    {
        size_t total = XH_N + WAT_N + WPT_N;
        int blocks = (int)((total + 255) / 256);
        prep_kernel<<<blocks, 256, 0, stream>>>(x, w_attn, w_proj, xh, wat, wpt);
    }
    // 2) QKV GEMM: 128 m-tiles * 48 n-tiles = 6144 waves / 8 per block
    qkv_gemm_kernel<<<768, 256, 0, stream>>>(xh, wat, b_attn, Qh, Kh, Vt);
    // 3) attention: B*H*(S/16) = 4096 waves / 4 per block
    attn_kernel<<<1024, 128, 0, stream>>>(Qh, Kh, Vt, ctxh);
    // 4) projection: 128 m-tiles * 16 n-tiles = 2048 waves / 8 per block
    proj_gemm_kernel<<<256, 256, 0, stream>>>(ctxh, wpt, b_proj, out);
}
